// ManifoldEncoder_10823317586024
// MI455X (gfx1250) — matrix operand order repaired and checked by reading, not compiled, not run on hardware
//
#include <hip/hip_runtime.h>
#include <hip/hip_bf16.h>
#include <math.h>

#define NPTS  2048
#define DIMF  784
#define KPAD  800
#define KNN1  6          // K+1 neighbors (includes self)
#define NCOMP 784
#define BIGV  1.0e6f
#define FWT   32
#define NITER 8

typedef __attribute__((ext_vector_type(16))) _Float16 v16h;
typedef __attribute__((ext_vector_type(8)))  _Float16 v8h;
typedef __attribute__((ext_vector_type(16))) __bf16   v16bf;
typedef __attribute__((ext_vector_type(8)))  __bf16   v8bf;
typedef __attribute__((ext_vector_type(8)))  float    v8f;
typedef __attribute__((ext_vector_type(4)))  unsigned int u32x4;
typedef __attribute__((ext_vector_type(8)))  int      i32x8;
typedef __attribute__((ext_vector_type(4)))  int      i32x4;

// ---------------- WMMA fragment traits ----------------
template<typename T> struct frag;
template<> struct frag<_Float16> {
  using vec16 = v16h; using vec8 = v8h;
  static __device__ inline v8f mma(vec16 a, vec16 b, v8f c) {
    return __builtin_amdgcn_wmma_f32_16x16x32_f16(false, a, false, b, (short)0, c, false, false);
  }
};
template<> struct frag<__bf16> {
  using vec16 = v16bf; using vec8 = v8bf;
  static __device__ inline v8f mma(vec16 a, vec16 b, v8f c) {
    return __builtin_amdgcn_wmma_f32_16x16x32_bf16(false, a, false, b, (short)0, c, false, false);
  }
};

__device__ inline float wave_sum32(float v) {
  #pragma unroll
  for (int o = 16; o > 0; o >>= 1) v += __shfl_xor(v, o, 32);
  return v;
}

__device__ inline unsigned rfl(unsigned x) {
  return (unsigned)__builtin_amdgcn_readfirstlane((int)x);
}

// ---------------- Tensor Data Mover: 2D strided tile -> LDS ----------------
// Elements are 2 bytes. LDS rows padded: 64B data + 16B pad => row stride 40 halves.
// tensor_dim* = remaining extent from tile origin (HW returns zeros for OOB reads).
__device__ inline void tdm_load_2d(unsigned lds_off, const void* gptr,
                                   unsigned rows_rem, unsigned cols_rem,
                                   unsigned stride_elems,
                                   unsigned tile_rows, unsigned tile_cols) {
  unsigned long long ga = (unsigned long long)(uintptr_t)gptr;
  unsigned w0g0 = rfl(1u);                                  // count=1, user mode
  unsigned w1g0 = rfl(lds_off);
  unsigned w2g0 = rfl((unsigned)(ga & 0xFFFFFFFFu));
  unsigned w3g0 = rfl((unsigned)((ga >> 32) & 0x01FFFFFFu) | (2u << 30)); // type=2 (image)
  u32x4 g0 = {w0g0, w1g0, w2g0, w3g0};

  unsigned c0 = (1u << 16)      // data_size = 2 bytes
              | (1u << 20)      // pad_enable
              | (3u << 22)      // pad_interval: 16 DWORDs (= one 64B row)
              | (3u << 25);     // pad_amount: 4 DWORDs (16B) -> stride 40 halves
  int w0 = (int)rfl(c0);
  int w1 = (int)rfl((cols_rem & 0xFFFFu) << 16);                      // tensor_dim0 lo
  int w2 = (int)rfl((cols_rem >> 16) | ((rows_rem & 0xFFFFu) << 16)); // dim0 hi | dim1 lo
  int w3 = (int)rfl((rows_rem >> 16) | (tile_cols << 16));            // dim1 hi | tile_dim0
  int w4 = (int)rfl(tile_rows & 0xFFFFu);                             // tile_dim1 (tile_dim2=0)
  int w5 = (int)rfl(stride_elems);                                    // tensor_dim0_stride lo
  i32x8 g1 = {w0, w1, w2, w3, w4, w5, 0, 0};
  i32x4 z4 = {0, 0, 0, 0};
  i32x8 z8 = {0, 0, 0, 0, 0, 0, 0, 0};
  __builtin_amdgcn_tensor_load_to_lds(g0, g1, z4, z4, z8, 0);
}

// ---------------- TDM + WMMA tiled GEMM ----------------
// C[M x N] = A[M x K] * B^T-layout:  A row-major (lda>=K), B stored n-major
// (B(k,n) = Bn[n*ldb + k], ldb>=K).  TC: store C transposed (C[n*ldc+m]).
// Block tile: 32(m) x 128(n); 8 waves as 2x4; each wave: 1 A-frag, 2 B-frags, 2 WMMA.
template<typename T, bool TC>
__global__ __launch_bounds__(256)
void mani_gemm(const T* __restrict__ A, int lda, const T* __restrict__ Bn, int ldb,
               float* __restrict__ C, int ldc, int M, int N, int Kd) {
  __shared__ __align__(16) T As[2][32 * 40];
  __shared__ __align__(16) T Bs[2][128 * 40];
  const int tid = threadIdx.x, lane = tid & 31, wave = tid >> 5;
  const int bm = blockIdx.y * 32, bn = blockIdx.x * 128;
  const int wm = (wave >> 2) * 16;        // 2 wave rows
  const int wn = (wave & 3) * 32;         // 4 wave cols, 32 wide each
  using vec16 = typename frag<T>::vec16;
  using vec8  = typename frag<T>::vec8;
  const int steps = Kd / 32;
  const unsigned rowsA = (unsigned)(M - bm);
  const unsigned rowsB = (unsigned)(N - bn);

  v8f acc0 = {}, acc1 = {};
  if (wave == 0) {
    tdm_load_2d((unsigned)(uintptr_t)&As[0][0], A + (size_t)bm * lda,
                rowsA, (unsigned)Kd, (unsigned)lda, 32u, 32u);
    tdm_load_2d((unsigned)(uintptr_t)&Bs[0][0], Bn + (size_t)bn * ldb,
                rowsB, (unsigned)Kd, (unsigned)ldb, 128u, 32u);
  }
  for (int i = 0; i < steps; ++i) {
    const int cur = i & 1;
    if (wave == 0) {
      if (i + 1 < steps) {
        const int nk = (i + 1) * 32;
        tdm_load_2d((unsigned)(uintptr_t)&As[cur ^ 1][0], A + (size_t)bm * lda + nk,
                    rowsA, (unsigned)(Kd - nk), (unsigned)lda, 32u, 32u);
        tdm_load_2d((unsigned)(uintptr_t)&Bs[cur ^ 1][0], Bn + (size_t)bn * ldb + nk,
                    rowsB, (unsigned)(Kd - nk), (unsigned)ldb, 128u, 32u);
        __builtin_amdgcn_s_wait_tensorcnt(2);   // current buffer's 2 DMAs done
      } else {
        __builtin_amdgcn_s_wait_tensorcnt(0);
      }
    }
    __syncthreads();
    // A fragment (16x32): lane<16 -> k {0..7,16..23}; lane>=16 -> +8
    const int mrow = wm + (lane & 15);
    const int ka = (lane >> 4) << 3;
    const T* ap = &As[cur][mrow * 40 + ka];
    vec8 alo = *(const vec8*)ap;
    vec8 ahi = *(const vec8*)(ap + 16);
    vec16 av = __builtin_shufflevector(alo, ahi, 0,1,2,3,4,5,6,7,8,9,10,11,12,13,14,15);
    // B fragments (32x16 each): lane<16 -> k {0..7,16..23}; lane>=16 -> +8
    const int kb = (lane >> 4) << 3;
    {
      const T* bp = &Bs[cur][(wn + (lane & 15)) * 40 + kb];
      vec8 blo = *(const vec8*)bp;
      vec8 bhi = *(const vec8*)(bp + 16);
      vec16 bv = __builtin_shufflevector(blo, bhi, 0,1,2,3,4,5,6,7,8,9,10,11,12,13,14,15);
      acc0 = frag<T>::mma(av, bv, acc0);
    }
    {
      const T* bp = &Bs[cur][(wn + 16 + (lane & 15)) * 40 + kb];
      vec8 blo = *(const vec8*)bp;
      vec8 bhi = *(const vec8*)(bp + 16);
      vec16 bv = __builtin_shufflevector(blo, bhi, 0,1,2,3,4,5,6,7,8,9,10,11,12,13,14,15);
      acc1 = frag<T>::mma(av, bv, acc1);
    }
    __syncthreads();
  }
  // C layout: VGPR r -> M = r + 8*lane[4], N = lane&15
  #pragma unroll
  for (int r = 0; r < 8; ++r) {
    int mi = bm + wm + r + ((lane >> 4) << 3);
    int n0 = bn + wn + (lane & 15);
    int n1 = n0 + 16;
    if (mi < M && n0 < N) { if (TC) C[(size_t)n0 * ldc + mi] = acc0[r]; else C[(size_t)mi * ldc + n0] = acc0[r]; }
    if (mi < M && n1 < N) { if (TC) C[(size_t)n1 * ldc + mi] = acc1[r]; else C[(size_t)mi * ldc + n1] = acc1[r]; }
  }
}

// ---------------- misc elementwise / reduction kernels ----------------
__global__ void mani_rownorm(const float* __restrict__ X, float* __restrict__ sq) {
  int lane = threadIdx.x & 31, wave = threadIdx.x >> 5;
  int row = blockIdx.x * 8 + wave;
  if (row >= NPTS) return;
  float s = 0.f;
  for (int k = lane; k < DIMF; k += 32) { float x = X[(size_t)row * DIMF + k]; s += x * x; }
  s = wave_sum32(s);
  if (lane == 0) sq[row] = s;
}

__global__ void mani_cvt_xh(const float* __restrict__ X, _Float16* __restrict__ Xh) {
  int i = blockIdx.x * blockDim.x + threadIdx.x;
  if (i >= NPTS * KPAD) return;
  int row = i / KPAD, k = i - row * KPAD;
  Xh[i] = (k < DIMF) ? (_Float16)X[(size_t)row * DIMF + k] : (_Float16)0.f;
}

__global__ void mani_cvt_bf16(const float* __restrict__ src, __bf16* __restrict__ dst, int n) {
  int i = blockIdx.x * blockDim.x + threadIdx.x;
  if (i < n) dst[i] = (__bf16)src[i];
}

__global__ void mani_dist(float* __restrict__ S, const float* __restrict__ sq) {
  int i = blockIdx.x * blockDim.x + threadIdx.x;
  if (i >= NPTS * NPTS) return;
  int r = i >> 11, c = i & (NPTS - 1);
  float d2 = sq[r] + sq[c] - 2.f * S[i];
  S[i] = sqrtf(fmaxf(d2, 0.f));
}

__global__ void mani_fill(float* __restrict__ G, float v, int n) {
  int i = blockIdx.x * blockDim.x + threadIdx.x;
  if (i < n) G[i] = v;
}

__global__ void mani_knn(const float* __restrict__ dist, float* __restrict__ G) {
  int row = blockIdx.x;          // one wave (32 threads) per row
  int lane = threadIdx.x;
  int   sel[KNN1];
  float selv[KNN1];
  for (int t = 0; t < KNN1; ++t) {
    float best = 3.0e38f; int bidx = NPTS;
    for (int j = lane; j < NPTS; j += 32) {
      bool used = false;
      for (int u = 0; u < t; ++u) used |= (sel[u] == j);
      float d = dist[(size_t)row * NPTS + j];
      if (!used && (d < best || (d == best && j < bidx))) { best = d; bidx = j; }
    }
    #pragma unroll
    for (int o = 16; o > 0; o >>= 1) {
      float ov = __shfl_xor(best, o, 32);
      int   oi = __shfl_xor(bidx, o, 32);
      if (ov < best || (ov == best && oi < bidx)) { best = ov; bidx = oi; }
    }
    sel[t] = bidx; selv[t] = best;
  }
  if (lane == 0) {
    for (int t = 0; t < KNN1; ++t) {
      int j = sel[t]; float v = selv[t];
      atomicMin((int*)&G[(size_t)row * NPTS + j], __float_as_int(v));
      atomicMin((int*)&G[(size_t)j * NPTS + row], __float_as_int(v));
    }
  }
}

__global__ void mani_diag0(float* __restrict__ G) {
  int i = blockIdx.x * blockDim.x + threadIdx.x;
  if (i < NPTS) G[(size_t)i * NPTS + i] = 0.f;
}

// ---------------- blocked Floyd-Warshall, 32x32 tiles ----------------
__global__ __launch_bounds__(1024) void mani_fw1(float* __restrict__ D, int kb) {
  __shared__ float t[FWT][FWT + 1];
  int tx = threadIdx.x, ty = threadIdx.y;
  size_t base = (size_t)(kb * FWT + ty) * NPTS + kb * FWT + tx;
  t[ty][tx] = D[base];
  __syncthreads();
  for (int k = 0; k < FWT; ++k) {
    float v = fminf(t[ty][tx], t[ty][k] + t[k][tx]);
    __syncthreads();
    t[ty][tx] = v;
    __syncthreads();
  }
  D[base] = t[ty][tx];
}

__global__ __launch_bounds__(1024) void mani_fw2(float* __restrict__ D, int kb) {
  __shared__ float piv[FWT][FWT + 1];
  __shared__ float cur[FWT][FWT + 1];
  int tx = threadIdx.x, ty = threadIdx.y;
  int t = blockIdx.x; if (t >= kb) t++;
  piv[ty][tx] = D[(size_t)(kb * FWT + ty) * NPTS + kb * FWT + tx];
  int r, c;
  if (blockIdx.y == 0) { r = kb * FWT + ty; c = t * FWT + tx; }
  else                 { r = t * FWT + ty;  c = kb * FWT + tx; }
  size_t base = (size_t)r * NPTS + c;
  cur[ty][tx] = D[base];
  __syncthreads();
  if (blockIdx.y == 0) {
    for (int k = 0; k < FWT; ++k) {
      float v = fminf(cur[ty][tx], piv[ty][k] + cur[k][tx]);
      __syncthreads();
      cur[ty][tx] = v;
      __syncthreads();
    }
  } else {
    for (int k = 0; k < FWT; ++k) {
      float v = fminf(cur[ty][tx], cur[ty][k] + piv[k][tx]);
      __syncthreads();
      cur[ty][tx] = v;
      __syncthreads();
    }
  }
  D[base] = cur[ty][tx];
}

__global__ __launch_bounds__(1024) void mani_fw3(float* __restrict__ D, int kb) {
  __shared__ float rowt[FWT][FWT + 1];
  __shared__ float colt[FWT][FWT + 1];
  int tx = threadIdx.x, ty = threadIdx.y;
  int bi = blockIdx.y; if (bi >= kb) bi++;
  int bj = blockIdx.x; if (bj >= kb) bj++;
  rowt[ty][tx] = D[(size_t)(bi * FWT + ty) * NPTS + kb * FWT + tx];
  colt[ty][tx] = D[(size_t)(kb * FWT + ty) * NPTS + bj * FWT + tx];
  __syncthreads();
  size_t base = (size_t)(bi * FWT + ty) * NPTS + bj * FWT + tx;
  float v = D[base];
  #pragma unroll 8
  for (int k = 0; k < FWT; ++k) v = fminf(v, rowt[ty][k] + colt[k][tx]);
  D[base] = v;
}

// ---------------- double centering ----------------
__global__ void mani_rowmean(const float* __restrict__ G, float* __restrict__ rm) {
  int lane = threadIdx.x & 31, wave = threadIdx.x >> 5;
  int row = blockIdx.x * 8 + wave;
  if (row >= NPTS) return;
  float s = 0.f;
  for (int j = lane; j < NPTS; j += 32) { float g = G[(size_t)row * NPTS + j]; s += g * g; }
  s = wave_sum32(s);
  if (lane == 0) rm[row] = s / (float)NPTS;
}

__global__ void mani_gmean(const float* __restrict__ rm, float* __restrict__ gm) {
  __shared__ float red[256];
  float s = 0.f;
  for (int i = threadIdx.x; i < NPTS; i += 256) s += rm[i];
  red[threadIdx.x] = s;
  __syncthreads();
  for (int o = 128; o > 0; o >>= 1) {
    if (threadIdx.x < o) red[threadIdx.x] += red[threadIdx.x + o];
    __syncthreads();
  }
  if (threadIdx.x == 0) gm[0] = red[0] / (float)NPTS;
}

__global__ void mani_bmat(float* __restrict__ Bm, const float* __restrict__ G,
                          const float* __restrict__ rm, const float* __restrict__ gm) {
  int i = blockIdx.x * blockDim.x + threadIdx.x;
  if (i >= NPTS * NPTS) return;
  int r = i >> 11, c = i & (NPTS - 1);
  float g = G[i];
  Bm[i] = -0.5f * (g * g - rm[r] - rm[c] + gm[0]);
}

// ---------------- subspace iteration pieces (V/W stored transposed [comp][point])
__global__ void mani_vinit(float* __restrict__ Vt, int n) {
  int i = blockIdx.x * blockDim.x + threadIdx.x;
  if (i >= n) return;
  unsigned x = (unsigned)i * 2654435761u;
  x ^= x >> 16; x *= 2246822519u; x ^= x >> 13; x *= 3266489917u; x ^= x >> 16;
  Vt[i] = ((float)(x & 0xFFFFu) / 32768.0f) - 1.0f;
}

__global__ void mani_chol(float* __restrict__ A, int C) {   // single block, in-place lower
  for (int j = 0; j < C; ++j) {
    if (threadIdx.x == 0) {
      float s = A[(size_t)j * C + j];
      for (int k = 0; k < j; ++k) { float l = A[(size_t)j * C + k]; s -= l * l; }
      A[(size_t)j * C + j] = sqrtf(fmaxf(s, 1e-12f));
    }
    __syncthreads();
    float dj = A[(size_t)j * C + j];
    for (int i = j + 1 + threadIdx.x; i < C; i += blockDim.x) {
      float s = A[(size_t)i * C + j];
      for (int k = 0; k < j; ++k) s -= A[(size_t)i * C + k] * A[(size_t)j * C + k];
      A[(size_t)i * C + j] = s / dj;
    }
    __syncthreads();
  }
}

// Vt = (W * L^{-T})^T : row-wise forward substitution; one wave per point r
__global__ __launch_bounds__(256) void mani_trsm(const float* __restrict__ Wt,
                                                 const float* __restrict__ L,
                                                 float* __restrict__ Vt) {
  __shared__ float xs[8 * NCOMP];
  int lane = threadIdx.x & 31, wave = threadIdx.x >> 5;
  int r = blockIdx.x * 8 + wave;
  float* x = &xs[wave * NCOMP];
  for (int j = lane; j < NCOMP; j += 32) x[j] = Wt[(size_t)j * NPTS + r];
  __syncthreads();
  for (int j = 0; j < NCOMP; ++j) {
    float s = 0.f;
    for (int k = lane; k < j; k += 32) s += x[k] * L[(size_t)j * NCOMP + k];
    s = wave_sum32(s);
    if (lane == 0) x[j] = (x[j] - s) / L[(size_t)j * NCOMP + j];
    __syncthreads();
  }
  for (int j = lane; j < NCOMP; j += 32) Vt[(size_t)j * NPTS + r] = x[j];
}

__global__ void mani_eigdot(const float* __restrict__ Vt, const float* __restrict__ Wt,
                            float* __restrict__ eig) {
  int lane = threadIdx.x & 31, wave = threadIdx.x >> 5;
  int col = blockIdx.x * 8 + wave;
  if (col >= NCOMP) return;
  float s = 0.f;
  for (int n = lane; n < NPTS; n += 32)
    s += Vt[(size_t)col * NPTS + n] * Wt[(size_t)col * NPTS + n];
  s = wave_sum32(s);
  if (lane == 0) eig[col] = s;
}

__global__ void mani_sign(const float* __restrict__ Vt, float* __restrict__ sgn) {
  int lane = threadIdx.x & 31, wave = threadIdx.x >> 5;
  int col = blockIdx.x * 8 + wave;
  if (col >= NCOMP) return;
  float best = -1.f; int bi = NPTS;
  for (int n = lane; n < NPTS; n += 32) {
    float a = fabsf(Vt[(size_t)col * NPTS + n]);
    if (a > best || (a == best && n < bi)) { best = a; bi = n; }
  }
  #pragma unroll
  for (int o = 16; o > 0; o >>= 1) {
    float ov = __shfl_xor(best, o, 32);
    int   oi = __shfl_xor(bi, o, 32);
    if (ov > best || (ov == best && oi < bi)) { best = ov; bi = oi; }
  }
  if (lane == 0) sgn[col] = (Vt[(size_t)col * NPTS + bi] >= 0.f) ? 1.f : -1.f;
}

__global__ void mani_rank(const float* __restrict__ eig, int* __restrict__ perm) {
  int c = blockIdx.x * blockDim.x + threadIdx.x;
  if (c >= NCOMP) return;
  float e = eig[c];
  int r = 0;
  for (int k = 0; k < NCOMP; ++k) {
    float ek = eig[k];
    if (ek > e || (ek == e && k < c)) r++;
  }
  perm[r] = c;
}

__global__ void mani_out(const float* __restrict__ Vt, const float* __restrict__ eig,
                         const float* __restrict__ sgn, const int* __restrict__ perm,
                         float* __restrict__ out) {
  int i = blockIdx.x * blockDim.x + threadIdx.x;
  if (i >= NPTS * NCOMP) return;
  int n = i / NCOMP, p = i - n * NCOMP;
  int q = perm[p];
  out[i] = sgn[q] * Vt[(size_t)q * NPTS + n] * sqrtf(fmaxf(eig[q], 0.f));
}

// ---------------- host orchestration ----------------
template<typename T, bool TC>
static void launch_gemm(const T* A, int lda, const T* Bn, int ldb, float* C, int ldc,
                        int M, int N, int Kd, hipStream_t s) {
  dim3 grid((N + 127) / 128, (M + 31) / 32);
  mani_gemm<T, TC><<<grid, 256, 0, s>>>(A, lda, Bn, ldb, C, ldc, M, N, Kd);
}

extern "C" void kernel_launch(void* const* d_in, const int* in_sizes, int n_in,
                              void* d_out, int out_size, void* d_ws, size_t ws_size,
                              hipStream_t stream) {
  (void)in_sizes; (void)n_in; (void)out_size; (void)ws_size;
  const float* X = (const float*)d_in[0];          // [2048, 784] f32
  float* out = (float*)d_out;                      // [2048, 784] f32

  char* w = (char*)d_ws;
  float*    M1  = (float*)(w);                                   // 16 MB: S -> dist -> Bmat
  float*    G   = (float*)(w + 16777216);                        // 16 MB: knn graph / geodesics
  __bf16*   Bbf = (__bf16*)G;                                    // reuse after G is dead
  _Float16* Xh  = (_Float16*)(w + 33554432);                     // 3.2 MB  [2048][800]
  float*    Vt  = (float*)(w + 36831232);                        // 6.4 MB  [784][2048]
  float*    Wt  = (float*)(w + 43253760);                        // 6.4 MB  [784][2048]
  __bf16*   Vtb = (__bf16*)(w + 49676288);                       // 3.2 MB
  __bf16*   Wtb = (__bf16*)(w + 52887552);                       // 3.2 MB
  float*    Gm  = (float*)(w + 56098816);                        // 2.5 MB  [784][784]
  float*    sq  = (float*)(w + 58557440);
  float*    rm  = sq + 2048;
  float*    gme = rm + 2048;
  float*    eig = gme + 64;
  float*    sgn = eig + 1024;
  int*      prm = (int*)(sgn + 1024);

  const int NN = NPTS * NPTS;
  const int NC = NPTS * NCOMP;
  const int EB = 256;

  // 1) squared row norms + f16 copy (K padded to 800)
  mani_rownorm<<<NPTS / 8, 256, 0, stream>>>(X, sq);
  mani_cvt_xh<<<(NPTS * KPAD + EB - 1) / EB, EB, 0, stream>>>(X, Xh);
  // 2) Gram via f16 WMMA + TDM: S = Xh * Xh^T  (Xh is both A and n-major B)
  launch_gemm<_Float16, false>(Xh, KPAD, Xh, KPAD, M1, NPTS, NPTS, NPTS, KPAD, stream);
  // 3) euclidean distances (in place)
  mani_dist<<<(NN + EB - 1) / EB, EB, 0, stream>>>(M1, sq);
  // 4) KNN graph
  mani_fill<<<(NN + EB - 1) / EB, EB, 0, stream>>>(G, BIGV, NN);
  mani_knn<<<NPTS, 32, 0, stream>>>(M1, G);
  mani_diag0<<<(NPTS + EB - 1) / EB, EB, 0, stream>>>(G);
  // 5) blocked Floyd-Warshall
  for (int kb = 0; kb < NPTS / FWT; ++kb) {
    mani_fw1<<<1, dim3(FWT, FWT), 0, stream>>>(G, kb);
    mani_fw2<<<dim3(NPTS / FWT - 1, 2), dim3(FWT, FWT), 0, stream>>>(G, kb);
    mani_fw3<<<dim3(NPTS / FWT - 1, NPTS / FWT - 1), dim3(FWT, FWT), 0, stream>>>(G, kb);
  }
  // 6) double-centered Gram B (symmetric => its n-major layout == row-major)
  mani_rowmean<<<NPTS / 8, 256, 0, stream>>>(G, rm);
  mani_gmean<<<1, 256, 0, stream>>>(rm, gme);
  mani_bmat<<<(NN + EB - 1) / EB, EB, 0, stream>>>(M1, G, rm, gme);
  // 7) bf16 copy of B (overwrites G, which is now dead)
  mani_cvt_bf16<<<(NN + EB - 1) / EB, EB, 0, stream>>>(M1, Bbf, NN);
  // 8) subspace iteration with Cholesky-QR (all GEMMs = bf16 WMMA + TDM)
  mani_vinit<<<(NC + EB - 1) / EB, EB, 0, stream>>>(Vt, NC);
  for (int it = 0; it < NITER; ++it) {
    mani_cvt_bf16<<<(NC + EB - 1) / EB, EB, 0, stream>>>(Vt, Vtb, NC);
    // W = B * V  -> stored transposed into Wt[comp][point]
    launch_gemm<__bf16, true>(Bbf, NPTS, Vtb, NPTS, Wt, NPTS, NPTS, NCOMP, NPTS, stream);
    mani_cvt_bf16<<<(NC + EB - 1) / EB, EB, 0, stream>>>(Wt, Wtb, NC);
    // Gm = W^T W  (A = Wt row-major, B = Wt n-major)
    launch_gemm<__bf16, false>(Wtb, NPTS, Wtb, NPTS, Gm, NCOMP, NCOMP, NCOMP, NPTS, stream);
    mani_chol<<<1, 256, 0, stream>>>(Gm, NCOMP);
    mani_trsm<<<NPTS / 8, 256, 0, stream>>>(Wt, Gm, Vt);
  }
  // 9) Rayleigh quotients, sign convention, descending order, embed
  mani_cvt_bf16<<<(NC + EB - 1) / EB, EB, 0, stream>>>(Vt, Vtb, NC);
  launch_gemm<__bf16, true>(Bbf, NPTS, Vtb, NPTS, Wt, NPTS, NPTS, NCOMP, NPTS, stream);
  mani_eigdot<<<(NCOMP + 7) / 8, 256, 0, stream>>>(Vt, Wt, eig);
  mani_sign<<<(NCOMP + 7) / 8, 256, 0, stream>>>(Vt, sgn);
  mani_rank<<<(NCOMP + EB - 1) / EB, EB, 0, stream>>>(eig, prm);
  mani_out<<<(NC + EB - 1) / EB, EB, 0, stream>>>(Vt, eig, sgn, prm, out);
}
